// WindowDecomposedMaSA1D_27144193311262
// MI455X (gfx1250) — matrix operand order repaired and checked by reading, not compiled, not run on hardware
//
#include <hip/hip_runtime.h>
#include <cmath>

// ---------------------------------------------------------------------------
// CDNA5 / gfx1250 implementation of windowed decay-masked attention + LCE.
// All GEMM-like stages run on V_WMMA_F32_16X16X32_BF16 (fp32 accumulate).
// GEMMs use 128x64 block tiles, 8 waves x (32x32 per-wave tile) = 4 WMMA/K-step.
// Workspace layout (bf16 elements), ~163 MiB total:
//   [0] x_bf16    (32768*512)  -> reused as g (gelu(dwconv)) after QKV
//   [1] q_bf16    (32768*512)  -> reused as S (attn + lce) after attn
//   [2] k_bf16    (32768*512)
//   [3] v_bf16    (32768*512)  (head-merged layout, also LCE input)
//   [4] attn_bf16 (32768*512)
//   [5] qkv_w bf16 (512*1536), pw_w bf16 (512*512), proj_w bf16 (512*512)
// ---------------------------------------------------------------------------

typedef __bf16 bf16_t;
typedef __bf16 v16bf __attribute__((ext_vector_type(16)));
typedef float  v8f   __attribute__((ext_vector_type(8)));

#define WMMA_BF16(a, b, c) \
  __builtin_amdgcn_wmma_f32_16x16x32_bf16(false, (a), false, (b), (short)0, (c), false, false)

// A fragment (16x32, M x K), contiguous-k layout (ldB_* use the same k order):
// lane<16: row=lane, elems 0..15 = K 0..15
// lane>=16: row=lane-16, elems 0..15 = K 16..31
static __device__ __forceinline__ v16bf ldA(const bf16_t* s, int lane, int m0, int k0, int ld) {
  const int m  = m0 + (lane & 15);
  const int kb = (lane >> 4) << 4;
  const bf16_t* p = s + m * ld + k0;
  v16bf a;
#pragma unroll
  for (int e = 0; e < 8; ++e) a[e] = p[kb + e];
#pragma unroll
  for (int e = 0; e < 8; ++e) a[8 + e] = p[kb + 8 + e];
  return a;
}

// B fragment (32x16, K x N) from a tile stored [n][k] (k contiguous):
// lane half selects K half (0..15 / 16..31), col = lane&15.
static __device__ __forceinline__ v16bf ldB_rows(const bf16_t* s, int lane, int n0, int k0, int ld) {
  const int n  = n0 + (lane & 15);
  const int kh = (lane >> 4) << 4;
  const bf16_t* p = s + n * ld + k0 + kh;
  v16bf b;
#pragma unroll
  for (int e = 0; e < 16; ++e) b[e] = p[e];
  return b;
}

// B fragment from a tile stored [k][n] (column gather).
static __device__ __forceinline__ v16bf ldB_cols(const bf16_t* s, int lane, int n0, int k0, int ld) {
  const int n  = n0 + (lane & 15);
  const int kh = k0 + ((lane >> 4) << 4);
  v16bf b;
#pragma unroll
  for (int e = 0; e < 16; ++e) b[e] = s[(kh + e) * ld + n];
  return b;
}

// ---------------------------------------------------------------------------
__global__ void cvt_f32_bf16(const float* __restrict__ s, bf16_t* __restrict__ d, int n) {
  int i = blockIdx.x * 256 + threadIdx.x;
  if (i < n) d[i] = (bf16_t)s[i];
}

// ---------------------------------------------------------------------------
// 128x64-tile GEMM, M rows via gridDim.y (BM=128), K fixed 512, bf16 in, f32 acc.
// 8 waves arranged 4(M) x 2(N); each wave owns a 32x32 tile -> 4 WMMAs / K-step.
// EPI 0: QKV  -> scatter q(*0.125)/k/v head-merged bf16, + qkv_b
// EPI 1: PW   -> out = bf16(acc + pw_b + attn)
// EPI 2: PROJ -> out = f32(acc + proj_b) * mask[row]
template <int EPI>
__global__ __launch_bounds__(256) void gemm128x64(
    const bf16_t* __restrict__ A, const bf16_t* __restrict__ Bw,
    const float* __restrict__ bias, const bf16_t* __restrict__ addb,
    const unsigned char* __restrict__ mask,
    bf16_t* __restrict__ oq, bf16_t* __restrict__ ok, bf16_t* __restrict__ ov,
    bf16_t* __restrict__ outb, float* __restrict__ outf, int N) {
  __shared__ bf16_t As[128 * 32];  // [m][k], 8 KB
  __shared__ bf16_t Bs[64 * 32];   // [n][k], 4 KB
  const int tid  = threadIdx.x;
  const int lane = tid & 31, wid = tid >> 5;
  const int bm = blockIdx.y, bn = blockIdx.x;
  const int wm = wid >> 1;        // 0..3 : wave row (32 rows each)
  const int wn = wid & 1;         // 0..1 : wave col (32 cols each)

  v8f acc00 = {0.f, 0.f, 0.f, 0.f, 0.f, 0.f, 0.f, 0.f};
  v8f acc01 = acc00, acc10 = acc00, acc11 = acc00;

  const int ar = tid >> 1, ac = (tid & 1) << 4;  // A: 128 rows x 2 chunks of 16
  const int br = tid >> 3, bc = (tid & 7) << 3;  // B: 32 k-rows x 8 chunks of 8
  const size_t arow = (size_t)(bm * 128 + ar) * 512;

  for (int kt = 0; kt < 512; kt += 32) {
    __syncthreads();
    {  // A tile: two 16B vector copies, stays [m][k]
      uint4 t0 = *(const uint4*)(A + arow + kt + ac);
      uint4 t1 = *(const uint4*)(A + arow + kt + ac + 8);
      *(uint4*)(&As[ar * 32 + ac])     = t0;
      *(uint4*)(&As[ar * 32 + ac + 8]) = t1;
    }
    {  // B tile: load [k][n] rows, transpose-scatter to [n][k]
      uint4 t = *(const uint4*)(Bw + (size_t)(kt + br) * N + bn * 64 + bc);
      const bf16_t* e = (const bf16_t*)&t;
#pragma unroll
      for (int j = 0; j < 8; ++j) Bs[(bc + j) * 32 + br] = e[j];
    }
    if (kt + 32 < 512) __builtin_prefetch(A + arow + kt + 32 + ac, 0, 1);  // global_prefetch_b8
    __syncthreads();

    v16bf a0 = ldA(As, lane, wm * 32, 0, 32);
    v16bf a1 = ldA(As, lane, wm * 32 + 16, 0, 32);
    v16bf b0 = ldB_rows(Bs, lane, wn * 32, 0, 32);
    v16bf b1 = ldB_rows(Bs, lane, wn * 32 + 16, 0, 32);
    acc00 = WMMA_BF16(a0, b0, acc00);
    acc01 = WMMA_BF16(a0, b1, acc01);
    acc10 = WMMA_BF16(a1, b0, acc10);
    acc11 = WMMA_BF16(a1, b1, acc11);
  }

#pragma unroll
  for (int ti = 0; ti < 2; ++ti) {
#pragma unroll
    for (int tj = 0; tj < 2; ++tj) {
      v8f acc = ti ? (tj ? acc11 : acc10) : (tj ? acc01 : acc00);
      const int nn = bn * 64 + wn * 32 + tj * 16 + (lane & 15);
#pragma unroll
      for (int j = 0; j < 8; ++j) {
        const int mm = bm * 128 + wm * 32 + ti * 16 + j + ((lane >> 4) << 3);
        float v = acc[j] + bias[nn];
        if (EPI == 0) {
          const int part = nn >> 9, c = nn & 511;
          const size_t di = ((size_t)mm << 9) + c;
          if (part == 0)      oq[di] = (bf16_t)(v * 0.125f);  // 1/sqrt(64)
          else if (part == 1) ok[di] = (bf16_t)v;
          else                ov[di] = (bf16_t)v;
        } else if (EPI == 1) {
          const size_t di = ((size_t)mm << 9) + nn;
          outb[di] = (bf16_t)(v + (float)addb[di]);
        } else {
          const size_t di = ((size_t)mm << 9) + nn;
          outf[di] = mask[mm] ? v : 0.f;
        }
      }
    }
  }
}

// ---------------------------------------------------------------------------
// One block per (window, head): S = (Q Kt) * gamma^|i-j|, mask, softmax, O = P V.
__global__ __launch_bounds__(256) void attn_win(
    const bf16_t* __restrict__ Q, const bf16_t* __restrict__ Km,
    const bf16_t* __restrict__ V, const float* __restrict__ gammas,
    const unsigned char* __restrict__ mask, bf16_t* __restrict__ O) {
  __shared__ bf16_t qs[64 * 64];
  __shared__ bf16_t ks[64 * 64];
  __shared__ bf16_t vs[64 * 64];
  __shared__ float  ss[64 * 64];
  __shared__ bf16_t ps[64 * 64];
  __shared__ float  dk[64];   // gamma^d decay table

  const int w = blockIdx.x >> 3, h = blockIdx.x & 7;
  const int tid = threadIdx.x, lane = tid & 31, wid = tid >> 5;

  if (tid < 64) dk[tid] = __expf(__logf(gammas[h]) * (float)tid);
  {  // cooperative 64x64 tile loads (head-merged layout, row stride 512)
    const int r = tid >> 2, c = (tid & 3) << 4;
    const size_t g = (size_t)(w * 64 + r) * 512 + h * 64 + c;
    *(uint4*)(&qs[r * 64 + c])     = *(const uint4*)(Q + g);
    *(uint4*)(&qs[r * 64 + c + 8]) = *(const uint4*)(Q + g + 8);
    *(uint4*)(&ks[r * 64 + c])     = *(const uint4*)(Km + g);
    *(uint4*)(&ks[r * 64 + c + 8]) = *(const uint4*)(Km + g + 8);
    *(uint4*)(&vs[r * 64 + c])     = *(const uint4*)(V + g);
    *(uint4*)(&vs[r * 64 + c + 8]) = *(const uint4*)(V + g + 8);
  }
  __syncthreads();

  const int tm = wid & 3, tn0 = (wid >> 2) << 1;

  // ---- S = Q K^T (K^T fragment == row-contiguous read of ks)
  v8f s0 = {0.f, 0.f, 0.f, 0.f, 0.f, 0.f, 0.f, 0.f};
  v8f s1 = s0;
#pragma unroll
  for (int kt = 0; kt < 64; kt += 32) {
    v16bf a  = ldA(qs, lane, tm * 16, kt, 64);
    v16bf b0 = ldB_rows(ks, lane, tn0 * 16, kt, 64);
    v16bf b1 = ldB_rows(ks, lane, (tn0 + 1) * 16, kt, 64);
    s0 = WMMA_BF16(a, b0, s0);
    s1 = WMMA_BF16(a, b1, s1);
  }

#pragma unroll
  for (int t = 0; t < 2; ++t) {
    v8f s = t ? s1 : s0;
    const int nn = (tn0 + t) * 16 + (lane & 15);
    const bool mk = mask[w * 64 + nn] != 0;
#pragma unroll
    for (int j = 0; j < 8; ++j) {
      const int mm = tm * 16 + j + ((lane >> 4) << 3);
      const int d = mm > nn ? (mm - nn) : (nn - mm);
      float val = s[j] * dk[d];  // gamma^|i-j| decay via LDS table
      if (!mk) val = -1e9f;
      ss[mm * 64 + nn] = val;
    }
  }
  __syncthreads();

  // ---- row softmax (one thread per query row)
  if (tid < 64) {
    float* row = &ss[tid * 64];
    float mx = -3.4e38f;
    for (int n = 0; n < 64; ++n) mx = fmaxf(mx, row[n]);
    float sum = 0.f;
    for (int n = 0; n < 64; ++n) { float e = __expf(row[n] - mx); row[n] = e; sum += e; }
    const float inv = 1.f / sum;
    for (int n = 0; n < 64; ++n) ps[tid * 64 + n] = (bf16_t)(row[n] * inv);
  }
  __syncthreads();

  // ---- O = P V (V fragment is a column gather of vs: [key][hd])
  v8f o0 = {0.f, 0.f, 0.f, 0.f, 0.f, 0.f, 0.f, 0.f};
  v8f o1 = o0;
#pragma unroll
  for (int kt = 0; kt < 64; kt += 32) {
    v16bf a  = ldA(ps, lane, tm * 16, kt, 64);
    v16bf b0 = ldB_cols(vs, lane, tn0 * 16, kt, 64);
    v16bf b1 = ldB_cols(vs, lane, (tn0 + 1) * 16, kt, 64);
    o0 = WMMA_BF16(a, b0, o0);
    o1 = WMMA_BF16(a, b1, o1);
  }
#pragma unroll
  for (int t = 0; t < 2; ++t) {
    v8f o = t ? o1 : o0;
    const int nn = (tn0 + t) * 16 + (lane & 15);
#pragma unroll
    for (int j = 0; j < 8; ++j) {
      const int mm = tm * 16 + j + ((lane >> 4) << 3);
      O[(size_t)(w * 64 + mm) * 512 + h * 64 + nn] = (bf16_t)o[j];
    }
  }
}

// ---------------------------------------------------------------------------
// Depthwise conv(5, SAME, per window) over v_merge + exact GELU -> bf16.
__global__ __launch_bounds__(256) void lce_dw_gelu(
    const bf16_t* __restrict__ V, const float* __restrict__ dwk, bf16_t* __restrict__ G) {
  const size_t i = (size_t)blockIdx.x * 256 + threadIdx.x;  // over 32768*512
  const int c = (int)(i & 511);
  const int row = (int)(i >> 9);
  const int t = row & 63;
  const int wbase = row - t;  // window start row
  float acc = 0.f;
#pragma unroll
  for (int j = 0; j < 5; ++j) {
    const int tt = t + j - 2;
    if (tt >= 0 && tt < 64)
      acc += (float)V[((size_t)(wbase + tt) << 9) + c] * dwk[j * 512 + c];
  }
  const float g = 0.5f * acc * (1.f + erff(acc * 0.70710678118654752f));
  G[i] = (bf16_t)g;
}

// ---------------------------------------------------------------------------
extern "C" void kernel_launch(void* const* d_in, const int* in_sizes, int n_in,
                              void* d_out, int out_size, void* d_ws, size_t ws_size,
                              hipStream_t stream) {
  (void)in_sizes; (void)n_in; (void)out_size; (void)ws_size;
  const float* x        = (const float*)d_in[0];
  const unsigned char* mask = (const unsigned char*)d_in[1];  // jnp bool_ = 1 byte
  const float* gammas   = (const float*)d_in[2];
  const float* qkv_w    = (const float*)d_in[3];
  const float* qkv_b    = (const float*)d_in[4];
  const float* proj_w   = (const float*)d_in[5];
  const float* proj_b   = (const float*)d_in[6];
  const float* dw_k     = (const float*)d_in[7];
  const float* pw_w     = (const float*)d_in[8];
  const float* pw_b     = (const float*)d_in[9];
  float* out = (float*)d_out;

  const size_t SZ = (size_t)32768 * 512;
  bf16_t* xb    = (bf16_t*)d_ws;       // later reused as g = gelu(dwconv)
  bf16_t* qb    = xb + SZ;             // later reused as S = attn + lce
  bf16_t* kb    = qb + SZ;
  bf16_t* vb    = kb + SZ;
  bf16_t* ab    = vb + SZ;             // attention output
  bf16_t* wqkv  = ab + SZ;
  bf16_t* wpw   = wqkv + (size_t)512 * 1536;
  bf16_t* wproj = wpw + (size_t)512 * 512;

  // 0) fp32 -> bf16 converts
  cvt_f32_bf16<<<(int)((SZ + 255) / 256), 256, 0, stream>>>(x, xb, (int)SZ);
  cvt_f32_bf16<<<(512 * 1536 + 255) / 256, 256, 0, stream>>>(qkv_w, wqkv, 512 * 1536);
  cvt_f32_bf16<<<(512 * 512 + 255) / 256, 256, 0, stream>>>(pw_w, wpw, 512 * 512);
  cvt_f32_bf16<<<(512 * 512 + 255) / 256, 256, 0, stream>>>(proj_w, wproj, 512 * 512);

  // 1) QKV projection (M=32768, N=1536, K=512) with q scaling + head scatter
  gemm128x64<0><<<dim3(1536 / 64, 32768 / 128), 256, 0, stream>>>(
      xb, wqkv, qkv_b, nullptr, nullptr, qb, kb, vb, nullptr, nullptr, 1536);

  // 2) windowed decay attention: 512 windows x 8 heads
  attn_win<<<4096, 256, 0, stream>>>(qb, kb, vb, gammas, mask, ab);

  // 3) LCE: depthwise conv + GELU (overwrites xb, no longer needed)
  lce_dw_gelu<<<(int)(SZ / 256), 256, 0, stream>>>(vb, dw_k, xb);

  // 4) pointwise GEMM + bias + attention add -> S (overwrites qb)
  gemm128x64<1><<<dim3(8, 256), 256, 0, stream>>>(
      xb, wpw, pw_b, ab, nullptr, nullptr, nullptr, nullptr, qb, nullptr, 512);

  // 5) output projection + bias + query mask -> fp32 d_out
  gemm128x64<2><<<dim3(8, 256), 256, 0, stream>>>(
      qb, wproj, proj_b, nullptr, mask, nullptr, nullptr, nullptr, nullptr, out, 512);
}
